// GATN_10539849744946
// MI455X (gfx1250) — matrix operand order repaired and checked by reading, not compiled, not run on hardware
//
#include <hip/hip_runtime.h>
#include <hip/hip_bf16.h>
#include <cstdint>

// ---------------- problem constants ----------------
#define NN      50000
#define NE      1600000
#define NG      512
#define IN_DIM  64
#define EMBED   128
#define NH      4
#define ND      32
#define HD      128      // NH*ND
#define D1      128
#define D2      64
#define D3      16
#define NEG_SLOPE 0.2f
#define NTILES  3125     // NN/16 exactly

// ---------------- workspace layout (float offsets) ----------------
#define OFF_FEAT 0              // NN*HD        = 6,400,000
#define OFF_EL   6400000        // NN*NH        = 200,000
#define OFF_ER   6600000        // NN*NH
#define OFF_EX   6800000        // NE*NH        = 6,400,000
#define OFF_MU   13200000       // NN*NH  (monotone-uint max)
#define OFF_DEN  13400000       // NN*NH  (zeroed)
#define OFF_RST  13600000       // NN*HD  (zeroed)
#define OFF_SUM  20000000       // NG*HD  (zeroed)
#define OFF_CNT  20065536       // NG     (zeroed)
#define OFF_HG   20066048       // NG*HD
#define OFF_O1   20131584       // NG*D1
#define OFF_O2   20197120       // NG*D2
#define ZERO_OFF OFF_DEN
#define ZERO_CNT (200000 + 6400000 + 65536 + 512)   // den+rst+sum+cnt contiguous

typedef __attribute__((ext_vector_type(16))) __bf16 v16bf;
typedef __attribute__((ext_vector_type(8)))  __bf16 v8bf;
typedef __attribute__((ext_vector_type(8)))  float  v8f;

// ---------------- WMMA fragment helpers (contiguous k = 16*(lane/16) + s order) ----------------
// A fragment 16x32 bf16 from row-major f32 (stride lda), K offset k0
__device__ __forceinline__ v16bf load_a_f32(const float* __restrict__ src, int lda,
                                            int k0, int lane) {
    int m  = lane & 15;
    int kg = (lane >> 4) << 4;                // 0 or 16
    const float* p = src + m * lda + k0 + kg;
    v16bf a;
#pragma unroll
    for (int i = 0; i < 8; ++i) a[i]     = (__bf16)p[i];        // K = kg+0..7
#pragma unroll
    for (int i = 0; i < 8; ++i) a[8 + i] = (__bf16)p[8 + i];    // K = kg+8..15
    return a;
}

// A fragment 16x32 from row-major bf16 LDS tile (stride HD) -> 2x ds_load_b128
__device__ __forceinline__ v16bf load_a_lds(const __bf16* __restrict__ s,
                                            int k0, int lane) {
    int m  = lane & 15;
    int kg = (lane >> 4) << 4;
    const __bf16* p = s + m * HD + k0 + kg;
    v8bf lo = *(const v8bf*)p;            // K = kg+0..7   (16B aligned)
    v8bf hi = *(const v8bf*)(p + 8);      // K = kg+8..15
    return __builtin_shufflevector(lo, hi, 0, 1, 2, 3, 4, 5, 6, 7,
                                           8, 9, 10, 11, 12, 13, 14, 15);
}

// B fragment 32x16 from LDS-transposed bf16 weights WT[n][k] (K-stride ldk)
// -> one 32B contiguous LDS read (2x ds_load_b128)
__device__ __forceinline__ v16bf load_b_ldsT(const __bf16* __restrict__ WT, int ldk,
                                             int k0, int col0, int lane) {
    int n  = lane & 15;
    int kb = (lane >> 4) << 4;                // 0 or 16
    return *(const v16bf*)(WT + (size_t)(col0 + n) * ldk + k0 + kb);
}

#define WMMA_BF16(a, b, c) \
    __builtin_amdgcn_wmma_f32_16x16x32_bf16(false, (a), false, (b), (short)0, (c), false, false)

// ---------------- fused node transform: feat = (feature@W0+b0)@Wfc ----------------
__global__ __launch_bounds__(128)
void k_node_transform(const float* __restrict__ feature,
                      const float* __restrict__ W0,
                      const float* __restrict__ b0,
                      const float* __restrict__ Wfc,
                      float* __restrict__ feat) {
    __shared__ __bf16 WfcT[HD * HD];          // 32 KB, [n][k] transposed bf16
    __shared__ __bf16 W0T[EMBED * IN_DIM];    // 16 KB, [n][k] transposed bf16
    __shared__ __bf16 hstage[4][16 * HD];     // 16 KB: per-wave C->A relayout stage
    // total static LDS = 64 KB

    // warm caches for the one-time weight pull (global_prefetch_b8 path)
    __builtin_prefetch(Wfc + threadIdx.x * 128, 0, 0);
    __builtin_prefetch(W0  + threadIdx.x * 64, 0, 0);

    // stage weights transposed as bf16 (coalesced global reads)
    for (int idx = threadIdx.x; idx < HD * HD; idx += blockDim.x) {
        int k = idx >> 7, n = idx & 127;
        WfcT[n * HD + k] = (__bf16)Wfc[idx];
    }
    for (int idx = threadIdx.x; idx < IN_DIM * EMBED; idx += blockDim.x) {
        int k = idx >> 7, n = idx & 127;
        W0T[n * IN_DIM + k] = (__bf16)W0[idx];
    }
    __syncthreads();

    int lane = threadIdx.x & 31;
    int wave = threadIdx.x >> 5;
    int tile = blockIdx.x * 4 + wave;
    if (tile > NTILES - 1) tile = NTILES - 1; // duplicate tail work (identical stores)
    int row0 = tile * 16;

    int n  = lane & 15;
    int mb = (lane >> 4) << 3;
    __bf16* hs = &hstage[wave][0];

    // GEMM0: h(16x128) = feature(16x64) @ W0(64x128) + b0
    v16bf a0 = load_a_f32(feature + (size_t)row0 * IN_DIM, IN_DIM, 0,  lane);
    v16bf a1 = load_a_f32(feature + (size_t)row0 * IN_DIM, IN_DIM, 32, lane);
#pragma unroll
    for (int ct = 0; ct < 8; ++ct) {
        int col0 = ct * 16;
        v8f c = {};
        v16bf bA = load_b_ldsT(W0T, IN_DIM, 0,  col0, lane);
        c = WMMA_BF16(a0, bA, c);
        v16bf bB = load_b_ldsT(W0T, IN_DIM, 32, col0, lane);
        c = WMMA_BF16(a1, bB, c);
        float bias = b0[col0 + n];
#pragma unroll
        for (int r = 0; r < 8; ++r)
            hs[(mb + r) * HD + col0 + n] = (__bf16)(c[r] + bias);
    }
    asm volatile("s_wait_dscnt 0" ::: "memory");   // within-wave LDS RAW fence

    // GEMM1: feat(16x128) = h(16x128) @ Wfc(128x128)
    v16bf ha[4];
#pragma unroll
    for (int kf = 0; kf < 4; ++kf) ha[kf] = load_a_lds(hs, kf * 32, lane);
#pragma unroll
    for (int ct = 0; ct < 8; ++ct) {
        int col0 = ct * 16;
        v8f c = {};
#pragma unroll
        for (int kf = 0; kf < 4; ++kf) {
            v16bf b = load_b_ldsT(WfcT, HD, kf * 32, col0, lane);
            c = WMMA_BF16(ha[kf], b, c);
        }
#pragma unroll
        for (int r = 0; r < 8; ++r)
            feat[(size_t)(row0 + mb + r) * HD + col0 + n] = c[r];
    }
}

// ---------------- attention scalars el/er ----------------
__global__ void k_elr(const float* __restrict__ feat,
                      const float* __restrict__ attn_l,
                      const float* __restrict__ attn_r,
                      float* __restrict__ el, float* __restrict__ er) {
    int gid = blockIdx.x * blockDim.x + threadIdx.x;
    if (gid >= NN * NH) return;
    int node = gid >> 2, hh = gid & 3;
    const float* f = feat + (size_t)node * HD + hh * ND;
    float sl = 0.f, sr = 0.f;
#pragma unroll
    for (int d = 0; d < ND; ++d) {
        float v = f[d];
        sl += v * attn_l[hh * ND + d];
        sr += v * attn_r[hh * ND + d];
    }
    el[gid] = sl; er[gid] = sr;
}

// ---------------- monotone float<->uint for atomicMax ----------------
__device__ __forceinline__ unsigned f2mono(float f) {
    unsigned u = __float_as_uint(f);
    return (u & 0x80000000u) ? ~u : (u | 0x80000000u);
}
__device__ __forceinline__ float mono2f(unsigned u) {
    return (u & 0x80000000u) ? __uint_as_float(u & 0x7fffffffu) : __uint_as_float(~u);
}

__global__ void k_init_max(unsigned* __restrict__ mu) {
    int gid = blockIdx.x * blockDim.x + threadIdx.x;
    if (gid < NN * NH) mu[gid] = f2mono(-1e30f);
}

__device__ __forceinline__ float edge_e(const float* el, const float* er,
                                        int s, int d, int hh) {
    float e = el[s * NH + hh] + er[d * NH + hh];
    return e > 0.f ? e : NEG_SLOPE * e;
}

__global__ void k_edge_max(const int* __restrict__ src, const int* __restrict__ dst,
                           const float* __restrict__ el, const float* __restrict__ er,
                           unsigned* __restrict__ mu) {
    int gid = blockIdx.x * blockDim.x + threadIdx.x;
    if (gid >= NE * NH) return;
    int e = gid >> 2, hh = gid & 3;
    float v = edge_e(el, er, src[e], dst[e], hh);
    atomicMax(&mu[dst[e] * NH + hh], f2mono(v));
}

__global__ void k_edge_exp(const int* __restrict__ src, const int* __restrict__ dst,
                           const float* __restrict__ el, const float* __restrict__ er,
                           const unsigned* __restrict__ mu,
                           float* __restrict__ ex, float* __restrict__ den) {
    int gid = blockIdx.x * blockDim.x + threadIdx.x;
    if (gid >= NE * NH) return;
    int e = gid >> 2, hh = gid & 3;
    int d = dst[e];
    float v = edge_e(el, er, src[e], d, hh);
    float x = __expf(v - mono2f(mu[d * NH + hh]));
    ex[gid] = x;
    atomicAdd(&den[d * NH + hh], x);
}

// ------- normalize + scatter messages (L2-bandwidth-bound kernel) -------
// 4 channels per thread: 1x b128 feat load + 4 f32 atomics, edge metadata amortized
__global__ void k_edge_aggr(const int* __restrict__ src, const int* __restrict__ dst,
                            const float* __restrict__ ex, const float* __restrict__ den,
                            const float* __restrict__ feat,
                            float* __restrict__ rst, float* __restrict__ atten) {
    long long gid = (long long)blockIdx.x * blockDim.x + threadIdx.x;
    if (gid >= (long long)NE * (HD / 4)) return;
    int e  = (int)(gid >> 5);
    int c  = ((int)gid & 31) * 4;       // channel group of 4
    int hh = c >> 5;
    int s = src[e], d = dst[e];
    float a = ex[e * NH + hh] / den[d * NH + hh];
    if ((c & 31) == 0) atten[e * NH + hh] = a;
    const float4 f = *(const float4*)&feat[(size_t)s * HD + c];
    float* r = &rst[(size_t)d * HD + c];
    atomicAdd(r + 0, a * f.x);
    atomicAdd(r + 1, a * f.y);
    atomicAdd(r + 2, a * f.z);
    atomicAdd(r + 3, a * f.w);
}

// ---------------- graph mean pooling ----------------
__global__ void k_pool_add(const float* __restrict__ rst, const float* __restrict__ gbias,
                           const int* __restrict__ gids,
                           float* __restrict__ sums, float* __restrict__ cnt) {
    int gid = blockIdx.x * blockDim.x + threadIdx.x;
    if (gid >= NN * HD) return;
    int node = gid >> 7, c = gid & 127;
    int g = gids[node];
    atomicAdd(&sums[g * HD + c], rst[gid] + gbias[c]);
    if (c == 0) atomicAdd(&cnt[g], 1.0f);
}

__device__ __forceinline__ float elu(float x) { return x > 0.f ? x : (__expf(x) - 1.f); }

__global__ void k_pool_fin(const float* __restrict__ sums, const float* __restrict__ cnt,
                           float* __restrict__ hg) {
    int gid = blockIdx.x * blockDim.x + threadIdx.x;
    if (gid >= NG * HD) return;
    int g = gid >> 7;
    float c = cnt[g];
    hg[gid] = elu(sums[gid] / fmaxf(c, 1.0f));
}

// ---------------- tiny MLP ----------------
__global__ void k_mlp1(const float* __restrict__ hg, const float* __restrict__ W,
                       const float* __restrict__ b, float* __restrict__ o) {
    int gid = blockIdx.x * blockDim.x + threadIdx.x;
    if (gid >= NG * D1) return;
    int g = gid / D1, j = gid % D1;
    float s = b[j];
#pragma unroll 8
    for (int k = 0; k < HD; ++k) s += hg[g * HD + k] * W[k * D1 + j];
    o[gid] = elu(s);
}
__global__ void k_mlp2(const float* __restrict__ i, const float* __restrict__ W,
                       const float* __restrict__ b, float* __restrict__ o) {
    int gid = blockIdx.x * blockDim.x + threadIdx.x;
    if (gid >= NG * D2) return;
    int g = gid / D2, j = gid % D2;
    float s = b[j];
#pragma unroll 8
    for (int k = 0; k < D1; ++k) s += i[g * D1 + k] * W[k * D2 + j];
    o[gid] = elu(s);
}
__global__ void k_mlp3(const float* __restrict__ i, const float* __restrict__ W,
                       const float* __restrict__ b, float* __restrict__ out) {
    int gid = blockIdx.x * blockDim.x + threadIdx.x;
    if (gid >= NG * D3) return;
    int g = gid / D3, j = gid % D3;
    float s = b[j];
#pragma unroll 8
    for (int k = 0; k < D2; ++k) s += i[g * D2 + k] * W[k * D3 + j];
    out[gid] = s;
}

// ---------------- launch ----------------
extern "C" void kernel_launch(void* const* d_in, const int* in_sizes, int n_in,
                              void* d_out, int out_size, void* d_ws, size_t ws_size,
                              hipStream_t stream) {
    const float* feature = (const float*)d_in[0];
    const int*   src     = (const int*)  d_in[1];
    const int*   dst     = (const int*)  d_in[2];
    const int*   gids    = (const int*)  d_in[3];
    const float* W0      = (const float*)d_in[4];
    const float* b0      = (const float*)d_in[5];
    const float* Wfc     = (const float*)d_in[6];
    const float* attn_l  = (const float*)d_in[7];
    const float* attn_r  = (const float*)d_in[8];
    const float* gbias   = (const float*)d_in[9];
    const float* W1      = (const float*)d_in[10];
    const float* b1      = (const float*)d_in[11];
    const float* W2      = (const float*)d_in[12];
    const float* b2      = (const float*)d_in[13];
    const float* W3      = (const float*)d_in[14];
    const float* b3      = (const float*)d_in[15];

    float* ws   = (float*)d_ws;
    float* out  = (float*)d_out;                 // [NG*D3] then atten [NE*NH]
    float* feat = ws + OFF_FEAT;
    float* el   = ws + OFF_EL;
    float* er   = ws + OFF_ER;
    float* ex   = ws + OFF_EX;
    unsigned* mu = (unsigned*)(ws + OFF_MU);
    float* den  = ws + OFF_DEN;
    float* rst  = ws + OFF_RST;
    float* sums = ws + OFF_SUM;
    float* cnt  = ws + OFF_CNT;
    float* hg   = ws + OFF_HG;
    float* o1   = ws + OFF_O1;
    float* o2   = ws + OFF_O2;

    // zero den/rst/sums/cnt (contiguous region), init max buffer
    hipMemsetAsync(ws + ZERO_OFF, 0, (size_t)ZERO_CNT * sizeof(float), stream);
    k_init_max<<<(NN * NH + 255) / 256, 256, 0, stream>>>(mu);

    // fused WMMA node transform (4 waves/block, one 16-row tile per wave)
    k_node_transform<<<(NTILES + 3) / 4, 128, 0, stream>>>(feature, W0, b0, Wfc, feat);

    k_elr<<<(NN * NH + 255) / 256, 256, 0, stream>>>(feat, attn_l, attn_r, el, er);

    k_edge_max<<<(NE * NH + 255) / 256, 256, 0, stream>>>(src, dst, el, er, mu);
    k_edge_exp<<<(NE * NH + 255) / 256, 256, 0, stream>>>(src, dst, el, er, mu, ex, den);

    long long aggr = (long long)NE * (HD / 4);
    k_edge_aggr<<<(unsigned)((aggr + 255) / 256), 256, 0, stream>>>(
        src, dst, ex, den, feat, rst, out + NG * D3);

    k_pool_add<<<(NN * HD + 255) / 256, 256, 0, stream>>>(rst, gbias, gids, sums, cnt);
    k_pool_fin<<<(NG * HD + 255) / 256, 256, 0, stream>>>(sums, cnt, hg);

    k_mlp1<<<(NG * D1 + 255) / 256, 256, 0, stream>>>(hg, W1, b1, o1);
    k_mlp2<<<(NG * D2 + 255) / 256, 256, 0, stream>>>(o1, W2, b2, o2);
    k_mlp3<<<(NG * D3 + 255) / 256, 256, 0, stream>>>(o2, W3, b3, out);
}
